// QuantumGraphNeuralNetwork_1786706395148
// MI455X (gfx1250) — matrix operand order repaired and checked by reading, not compiled, not run on hardware
//
#include <hip/hip_runtime.h>
#include <math.h>

#define NQ 11
#define NL 3
#define SDIM 2048
#define BATCH 8
#define ATOMS 96
#define NNODE 768
#define FDIM 64

typedef __attribute__((ext_vector_type(16))) _Float16 v16h;
typedef __attribute__((ext_vector_type(8)))  float    v8f;

// ---------------------------------------------------------------------------
// Kernel 0: zero the pooled accumulator R[8][2048]
// ---------------------------------------------------------------------------
__global__ void zero_kernel(float* __restrict__ p, int n) {
    int i = blockIdx.x * blockDim.x + threadIdx.x;
    if (i < n) p[i] = 0.0f;
}

// ---------------------------------------------------------------------------
// Kernel 1: fused feature MLP via WMMA.
//   h = relu(nf @ W1[:, :64]^T + b1)   (padded x cols 64..99 are zero)
//   f = tanh(h @ W2^T + b2)            (N=11 padded to 16)
// 6 blocks x 256 threads (8 waves); each wave owns one 16-row tile.
// ---------------------------------------------------------------------------
__global__ __launch_bounds__(256) void mlp_f_kernel(
    const float* __restrict__ nf, const float* __restrict__ W1,
    const float* __restrict__ b1, const float* __restrict__ W2,
    const float* __restrict__ b2, float* __restrict__ f_out)
{
    __shared__ _Float16 hbuf[8][16][64];   // per-wave 16x64 f16 tile of h
    const int lane = threadIdx.x & 31;
    const int wave = threadIdx.x >> 5;
    const int half = lane >> 4;            // wave32 lane group
    const int m    = lane & 15;
    const int rt   = blockIdx.x * 8 + wave;  // row tile 0..47
    const int row0 = rt * 16;

    // A fragments (16x32 f16 layout): e<8 -> K=8*half+e ; e>=8 -> K=16+8*half+(e-8)
    v16h a0, a1;
    #pragma unroll
    for (int e = 0; e < 16; ++e) {
        int kk = (e < 8) ? (8 * half + e) : (16 + 8 * half + (e - 8));
        a0[e] = (_Float16)nf[(row0 + m) * FDIM + kk];
        a1[e] = (_Float16)nf[(row0 + m) * FDIM + 32 + kk];
    }

    // GEMM1: 4 N-tiles of 16 output features, K = 2 chunks of 32
    #pragma unroll
    for (int nt = 0; nt < 4; ++nt) {
        v8f c = {};
        #pragma unroll
        for (int kc = 0; kc < 2; ++kc) {
            v16h bfrag;  // B (32x16 f16): e<8 -> K = kc*32 + 8*half + e ; e>=8 -> K = kc*32 + 16 + 8*half + (e-8), N = m
            #pragma unroll
            for (int e = 0; e < 16; ++e) {
                int kk = kc * 32 + ((e < 8) ? (8 * half + e) : (16 + 8 * half + (e - 8)));
                bfrag[e] = (_Float16)W1[(nt * 16 + m) * 100 + kk];
            }
            c = __builtin_amdgcn_wmma_f32_16x16x32_f16(
                    false, (kc == 0) ? a0 : a1, false, bfrag,
                    (short)0, c, false, false);
        }
        // C layout: lane m -> N, VGPR r -> M = 8*half + r ; add bias, relu, stash f16
        #pragma unroll
        for (int r = 0; r < 8; ++r) {
            int mm = half * 8 + r;
            int n  = nt * 16 + m;
            float hv = c[r] + b1[n];
            hbuf[wave][mm][n] = (_Float16)(hv > 0.0f ? hv : 0.0f);
        }
    }
    __syncthreads();

    // GEMM2: A = h tile from LDS, B = W2^T (11 valid cols, zero-padded to 16)
    v16h a2k0, a2k1;
    #pragma unroll
    for (int e = 0; e < 16; ++e) {
        int kk = (e < 8) ? (8 * half + e) : (16 + 8 * half + (e - 8));
        a2k0[e] = hbuf[wave][m][kk];
        a2k1[e] = hbuf[wave][m][32 + kk];
    }
    v8f c2 = {};
    #pragma unroll
    for (int kc = 0; kc < 2; ++kc) {
        v16h bfrag;
        #pragma unroll
        for (int e = 0; e < 16; ++e) {
            int kk = kc * 32 + ((e < 8) ? (8 * half + e) : (16 + 8 * half + (e - 8)));
            bfrag[e] = (m < NQ) ? (_Float16)W2[m * 64 + kk] : (_Float16)0.0f;
        }
        c2 = __builtin_amdgcn_wmma_f32_16x16x32_f16(
                 false, (kc == 0) ? a2k0 : a2k1, false, bfrag,
                 (short)0, c2, false, false);
    }
    if (m < NQ) {
        #pragma unroll
        for (int r = 0; r < 8; ++r) {
            int mm = half * 8 + r;
            f_out[(row0 + mm) * NQ + m] = tanhf(c2[r] + b2[m]);
        }
    }
}

// ---------------------------------------------------------------------------
// Kernel 2: quantum circuit. One block per sample; the 2048-float statevector
// lives in LDS for all 63 gates (amplitudes stay REAL through the circuit).
// Finally fold w[a%11]*amps into the batch accumulator R via f32 atomics.
// ---------------------------------------------------------------------------
__global__ __launch_bounds__(256) void circuit_kernel(
    const float* __restrict__ f, const float* __restrict__ rot,
    const float* __restrict__ ent, const float* __restrict__ pool_w,
    float* __restrict__ R)
{
    __shared__ float amps[SDIM];
    const int n   = blockIdx.x;
    const int b   = n / ATOMS;
    const int a   = n - b * ATOMS;
    const int tid = threadIdx.x;

    for (int i = tid; i < SDIM; i += 256) amps[i] = (i == 0) ? 1.0f : 0.0f;

    float fq[NQ];
    #pragma unroll
    for (int q = 0; q < NQ; ++q) fq[q] = f[n * NQ + q];
    __syncthreads();

    for (int l = 0; l < NL; ++l) {
        // single-qubit rotations (real 2x2 per the torch construction)
        for (int q = 0; q < NQ; ++q) {
            float rx = rot[(l * NQ + q) * 3 + 0] * fq[q] * 0.5f;
            float ry = rot[(l * NQ + q) * 3 + 1] * fq[q] * 0.5f;
            float rz = rot[(l * NQ + q) * 3 + 2] * fq[q] * 0.5f;
            float cx = __cosf(rx), sx = __sinf(rx);
            float cy = __cosf(ry), sy = __sinf(ry);
            float cz = __cosf(rz), sz = __sinf(rz);
            float m00 =  cx * cy * cz, m01 = -sx * sy * sz;
            float m10 =  sx * sy * cz, m11 =  cx * cy * sz;
            int lomask = (1 << q) - 1;
            for (int p = tid; p < SDIM / 2; p += 256) {
                int i0 = ((p >> q) << (q + 1)) | (p & lomask);
                int i1 = i0 | (1 << q);
                float v0 = amps[i0], v1 = amps[i1];
                amps[i0] = m00 * v0 + m01 * v1;
                amps[i1] = m10 * v0 + m11 * v1;
            }
            __syncthreads();
        }
        // parameterized CNOT mixing: pairs with ctrl bit q == 1, flip bit q+1
        for (int q = 0; q < NQ - 1; ++q) {
            float pm = 1.0f / (1.0f + __expf(-ent[l * (NQ - 1) + q]));
            int lomask = (1 << q) - 1;
            for (int j = tid; j < SDIM / 4; j += 256) {
                int s0 = ((j >> q) << (q + 2)) | (1 << q) | (j & lomask);
                int s1 = s0 | (1 << (q + 1));
                float v0 = amps[s0], v1 = amps[s1];
                amps[s0] = (1.0f - pm) * v0 + pm * v1;
                amps[s1] = (1.0f - pm) * v1 + pm * v0;
            }
            __syncthreads();
        }
    }

    float w = 1.0f / (1.0f + __expf(-pool_w[a % NQ]));
    for (int s = tid; s < SDIM; s += 256)
        atomicAdd(&R[b * SDIM + s], w * amps[s]);
}

// ---------------------------------------------------------------------------
// Kernel 3: per-batch tail. Phase phi(s) = bits(s) . theta_total, then
// norm / means -> rep = [rb x11, ib x11] -> MLP 22->256->128->64.
// ---------------------------------------------------------------------------
__global__ __launch_bounds__(256) void tail_kernel(
    const float* __restrict__ R, const float* __restrict__ msg,
    const float* __restrict__ Wo1, const float* __restrict__ bo1,
    const float* __restrict__ Wo2, const float* __restrict__ bo2,
    const float* __restrict__ Wo3, const float* __restrict__ bo3,
    float* __restrict__ out)
{
    __shared__ float red0[256], red1[256], red2[256];
    __shared__ float h1s[256];
    __shared__ float h2s[128];
    __shared__ float rep2[2];
    const int b   = blockIdx.x;
    const int tid = threadIdx.x;

    float theta[NQ];
    #pragma unroll
    for (int q = 0; q < NQ; ++q) {
        float t = 0.0f;
        #pragma unroll
        for (int l = 0; l < NL; ++l)
            #pragma unroll
            for (int c = 0; c < 3; ++c)
                t += msg[(l * NQ + q) * 3 + c];
        theta[q] = t;
    }

    float acc2 = 0.0f, accc = 0.0f, accs = 0.0f;
    for (int s = tid; s < SDIM; s += 256) {
        float r = R[b * SDIM + s];
        float phi = 0.0f;
        #pragma unroll
        for (int q = 0; q < NQ; ++q) phi += ((s >> q) & 1) ? theta[q] : 0.0f;
        acc2 += r * r;
        accc += r * __cosf(phi);
        accs += r * __sinf(phi);
    }
    red0[tid] = acc2; red1[tid] = accc; red2[tid] = accs;
    __syncthreads();
    for (int st = 128; st > 0; st >>= 1) {
        if (tid < st) {
            red0[tid] += red0[tid + st];
            red1[tid] += red1[tid + st];
            red2[tid] += red2[tid + st];
        }
        __syncthreads();
    }
    if (tid == 0) {
        float norm = fmaxf(sqrtf(red0[0]), 1e-12f);
        rep2[0] = (red1[0] / (float)SDIM) / norm;   // rb
        rep2[1] = (red2[0] / (float)SDIM) / norm;   // ib
    }
    __syncthreads();
    float rb = rep2[0], ib = rep2[1];

    // h1 = relu(Wo1 @ rep + bo1), rep[k] = k<11 ? rb : ib
    {
        float sA = 0.0f, sB = 0.0f;
        #pragma unroll
        for (int k = 0; k < NQ; ++k)      sA += Wo1[tid * 22 + k];
        #pragma unroll
        for (int k = NQ; k < 2 * NQ; ++k) sB += Wo1[tid * 22 + k];
        h1s[tid] = fmaxf(rb * sA + ib * sB + bo1[tid], 0.0f);
    }
    __syncthreads();
    if (tid < 128) {
        float acc = bo2[tid];
        for (int k = 0; k < 256; ++k) acc += Wo2[tid * 256 + k] * h1s[k];
        h2s[tid] = fmaxf(acc, 0.0f);
    }
    __syncthreads();
    if (tid < 64) {
        float acc = bo3[tid];
        for (int k = 0; k < 128; ++k) acc += Wo3[tid * 128 + k] * h2s[k];
        out[b * 64 + tid] = acc;
    }
}

// ---------------------------------------------------------------------------
extern "C" void kernel_launch(void* const* d_in, const int* in_sizes, int n_in,
                              void* d_out, int out_size, void* d_ws, size_t ws_size,
                              hipStream_t stream) {
    (void)in_sizes; (void)n_in; (void)out_size; (void)ws_size;
    const float* nf    = (const float*)d_in[0];
    // d_in[1] edge_indices: unused by the math
    const float* W1    = (const float*)d_in[2];
    const float* b1    = (const float*)d_in[3];
    const float* W2    = (const float*)d_in[4];
    const float* b2    = (const float*)d_in[5];
    const float* rot   = (const float*)d_in[6];
    const float* ent   = (const float*)d_in[7];
    const float* msg   = (const float*)d_in[8];
    const float* poolw = (const float*)d_in[9];
    const float* Wo1   = (const float*)d_in[10];
    const float* bo1   = (const float*)d_in[11];
    const float* Wo2   = (const float*)d_in[12];
    const float* bo2   = (const float*)d_in[13];
    const float* Wo3   = (const float*)d_in[14];
    const float* bo3   = (const float*)d_in[15];

    float* f_ws = (float*)d_ws;               // 768*11 floats
    float* R    = f_ws + NNODE * NQ;          // 8*2048 floats

    zero_kernel   <<<(BATCH * SDIM + 255) / 256, 256, 0, stream>>>(R, BATCH * SDIM);
    mlp_f_kernel  <<<NNODE / (16 * 8), 256, 0, stream>>>(nf, W1, b1, W2, b2, f_ws);
    circuit_kernel<<<NNODE, 256, 0, stream>>>(f_ws, rot, ent, poolw, R);
    tail_kernel   <<<BATCH, 256, 0, stream>>>(R, msg, Wo1, bo1, Wo2, bo2, Wo3, bo3,
                                              (float*)d_out);
}
